// SoftmaxApprSamplerUniform_8254927143135
// MI455X (gfx1250) — matrix operand order repaired and checked by reading, not compiled, not run on hardware
//
#include <hip/hip_runtime.h>
#include <hip/hip_bf16.h>
#include <stdint.h>

// ---------------------------------------------------------------------------
// Fused SoftmaxApprSampler for MI455X (gfx1250, wave32, WMMA 16x16x32 f16).
//
// Inputs (setup_inputs order):
//  0 query   [B,128] f32      6 c0_     [K+1,64] f32
//  1 pos_items [B] int        7 c1_     [K+1,64] f32
//  2 num_neg [1] int          8 cd0     [N+1] int
//  3 c0      [K,64] f32       9 cd1     [N+1] int
//  4 c1      [K,64] f32      10 indices [N]   int
//  5 wkk     [K,K]  f32      11 indptr  [K*K+1] int
//
// Output (flat float): pos_prop[B] | neg_items[B*nneg] | neg_prob[B*nneg]
// ---------------------------------------------------------------------------

typedef __attribute__((ext_vector_type(16))) _Float16 v16h;
typedef __attribute__((ext_vector_type(8)))  float    v8f;

#define KC     64     // clusters per half
#define DHALF  64     // D/2
#define ROWS_PER_WAVE 16
#define WAVES_PER_BLK 2
#define ROWS_PER_BLK  (ROWS_PER_WAVE * WAVES_PER_BLK)
#define LPITCH 68     // LDS row pitch for per-wave arrays
#define WPITCH 65     // LDS pitch for wkk (odd -> conflict-free row gather)

// splitmix-style hash -> u32
static __device__ __forceinline__ uint32_t hash3(uint32_t a, uint32_t b, uint32_t c) {
  uint32_t x = a * 0x9E3779B1u + b * 0x85EBCA77u + c * 0xC2B2AE3Du;
  x ^= x >> 16; x *= 0x7FEB352Du;
  x ^= x >> 15; x *= 0x846CA68Bu;
  x ^= x >> 16;
  return x;
}
static __device__ __forceinline__ float u01(uint32_t h) {
  return (float)(h >> 8) * (1.0f / 16777216.0f);  // [0,1)
}

// A-fragment (16x32 f16, MxK). rowbase -> K=0 of this row/tile.
// Lane holds K in [16h..16h+15], h = lane>>4 (matches load_bfrag_T).
static __device__ __forceinline__ v16h load_afrag(const float* rowbase, int half) {
  v16h a;
  const float* p0 = rowbase + 16 * half;
  const float* p1 = rowbase + 16 * half + 8;
#pragma unroll
  for (int i = 0; i < 8; ++i) a[i] = (_Float16)p0[i];
#pragma unroll
  for (int i = 0; i < 8; ++i) a[8 + i] = (_Float16)p1[i];
  return a;
}

// B-fragment (32x16 f16, KxN) with B[k][n] = src[n][k] (B = src^T); src rows
// contiguous f32. Lane n = lane&15 holds K in [16h..16h+15].
static __device__ __forceinline__ v16h load_bfrag_T(const float* src_row_kbase, int half) {
  v16h b;
  const float* p = src_row_kbase + 16 * half;
#pragma unroll
  for (int i = 0; i < 16; ++i) b[i] = (_Float16)p[i];
  return b;
}

__global__ __launch_bounds__(WAVES_PER_BLK * 32)
void sampler_fused_kernel(const float* __restrict__ query,
                          const float* __restrict__ c0,
                          const float* __restrict__ c1,
                          const float* __restrict__ wkk,
                          const int*   __restrict__ indices,
                          const int*   __restrict__ indptr,
                          const int*   __restrict__ nneg_ptr,
                          float* __restrict__ out, int B) {
  __shared__ float sh_r0 [WAVES_PER_BLK][ROWS_PER_WAVE][LPITCH];  // raw logits 0
  __shared__ float sh_r1 [WAVES_PER_BLK][ROWS_PER_WAVE][LPITCH];  // raw logits 1
  __shared__ float sh_r1s[WAVES_PER_BLK][ROWS_PER_WAVE][LPITCH];  // softmax(r1)
  __shared__ float sh_cdf[WAVES_PER_BLK][ROWS_PER_WAVE][LPITCH];  // cumsum(s0)
  __shared__ float sh_t1 [WAVES_PER_BLK][ROWS_PER_WAVE][LPITCH];  // r1s@wkk^T (stage-2 normalizers)
  __shared__ float sh_wkk[KC][WPITCH];                            // staged wkk
  __shared__ float sh_max0[WAVES_PER_BLK][ROWS_PER_WAVE];
  __shared__ float sh_sum0[WAVES_PER_BLK][ROWS_PER_WAVE];

  const int tid  = threadIdx.x;
  const int w    = tid >> 5;
  const int lane = tid & 31;
  const int half = lane >> 4;
  const int mrow = lane & 15;
  const int m0   = blockIdx.x * ROWS_PER_BLK + w * ROWS_PER_WAVE;
  const int nneg = *nneg_ptr;

  __builtin_prefetch(indptr, 0, 3);

  // ---- Stage wkk into LDS (64 threads x 64 elements) ----
#pragma unroll
  for (int i = 0; i < (KC * KC) / (WAVES_PER_BLK * 32); ++i) {
    int e = tid + i * (WAVES_PER_BLK * 32);
    sh_wkk[e >> 6][e & 63] = wkk[e];
  }

  // ---- Phase 1: r0 = q0 @ c0^T, r1 = q1 @ c1^T via WMMA f16 (f32 acc) ----
  const float* qrow = query + (size_t)(m0 + mrow) * (2 * DHALF);
  v16h a0t0 = load_afrag(qrow + 0,          half);
  v16h a0t1 = load_afrag(qrow + 32,         half);
  v16h a1t0 = load_afrag(qrow + DHALF + 0,  half);
  v16h a1t1 = load_afrag(qrow + DHALF + 32, half);

#pragma unroll
  for (int nt = 0; nt < KC / 16; ++nt) {
    const float* c0row = c0 + (size_t)(nt * 16 + mrow) * DHALF;
    const float* c1row = c1 + (size_t)(nt * 16 + mrow) * DHALF;
    v8f acc0 = {};
    v8f acc1 = {};
    acc0 = __builtin_amdgcn_wmma_f32_16x16x32_f16(false, a0t0, false,
            load_bfrag_T(c0row + 0,  half), (short)0, acc0, false, false);
    acc0 = __builtin_amdgcn_wmma_f32_16x16x32_f16(false, a0t1, false,
            load_bfrag_T(c0row + 32, half), (short)0, acc0, false, false);
    acc1 = __builtin_amdgcn_wmma_f32_16x16x32_f16(false, a1t0, false,
            load_bfrag_T(c1row + 0,  half), (short)0, acc1, false, false);
    acc1 = __builtin_amdgcn_wmma_f32_16x16x32_f16(false, a1t1, false,
            load_bfrag_T(c1row + 32, half), (short)0, acc1, false, false);
    // C layout: lane holds column n = lane&15, rows M = r + 8*half
#pragma unroll
    for (int r = 0; r < 8; ++r) {
      sh_r0[w][r + 8 * half][nt * 16 + mrow] = acc0[r];
      sh_r1[w][r + 8 * half][nt * 16 + mrow] = acc1[r];
    }
  }
  __syncthreads();

  // ---- Phase 2: row softmax stats; materialize r1s ----
  if (lane < ROWS_PER_WAVE) {
    const int row = lane;
    float mx0 = -3.0e38f, mx1 = -3.0e38f;
#pragma unroll 8
    for (int k = 0; k < KC; ++k) {
      mx0 = fmaxf(mx0, sh_r0[w][row][k]);
      mx1 = fmaxf(mx1, sh_r1[w][row][k]);
    }
    float s0 = 0.f, s1 = 0.f;
#pragma unroll 8
    for (int k = 0; k < KC; ++k) {
      s0 += __expf(sh_r0[w][row][k] - mx0);
      float e1 = __expf(sh_r1[w][row][k] - mx1);
      sh_r1s[w][row][k] = e1;
      s1 += e1;
    }
    float inv1 = 1.0f / s1;
#pragma unroll 8
    for (int k = 0; k < KC; ++k) sh_r1s[w][row][k] *= inv1;
    sh_max0[w][row] = mx0;
    sh_sum0[w][row] = s0;
  }
  __syncthreads();

  // ---- Phase 3: s0pre = r1s @ wkk^T via WMMA (B frags from LDS wkk).
  //      Keep s0pre (stage-2 normalizers) AND s0 = s0pre * r0s. ----
  const float* r1srow = &sh_r1s[w][mrow][0];
  v16h as0 = load_afrag(r1srow + 0,  half);
  v16h as1 = load_afrag(r1srow + 32, half);
#pragma unroll
  for (int nt = 0; nt < KC / 16; ++nt) {
    const float* wrow = &sh_wkk[nt * 16 + mrow][0];   // B = wkk^T
    v8f acc = {};
    acc = __builtin_amdgcn_wmma_f32_16x16x32_f16(false, as0, false,
            load_bfrag_T(wrow + 0,  half), (short)0, acc, false, false);
    acc = __builtin_amdgcn_wmma_f32_16x16x32_f16(false, as1, false,
            load_bfrag_T(wrow + 32, half), (short)0, acc, false, false);
#pragma unroll
    for (int r = 0; r < 8; ++r) {
      int mm  = r + 8 * half;
      int col = nt * 16 + mrow;
      float r0s = __expf(sh_r0[w][mm][col] - sh_max0[w][mm]) / sh_sum0[w][mm];
      sh_t1 [w][mm][col] = acc[r];          // tot1 for (row=mm, k0=col)
      sh_cdf[w][mm][col] = acc[r] * r0s;    // unnormalized s0
    }
  }
  __syncthreads();

  // ---- Phase 4: inclusive prefix sum (CDF) per row ----
  if (lane < ROWS_PER_WAVE) {
    float c = 0.f;
#pragma unroll 8
    for (int k = 0; k < KC; ++k) { c += sh_cdf[w][lane][k]; sh_cdf[w][lane][k] = c; }
  }
  __syncthreads();

  // ---- Phase 5: hierarchical sampling (2 streams x 16 rows per wave) ----
  {
    const int row    = lane & 15;
    const int stream = lane >> 4;
    const int grow   = m0 + row;
    const float total0 = sh_cdf[w][row][KC - 1];
    float* neg_items = out + B;
    float* neg_prob  = out + B + (size_t)B * nneg;

    for (int j = stream; j < nneg; j += 2) {
      // stage 1: k0 ~ Categorical(s0) via CDF scan
      float u = u01(hash3((uint32_t)grow, (uint32_t)j, 0x0000002Au)) * total0;
      int k0 = KC - 1;
      for (int k = 0; k < KC; ++k) {
        if (u <= sh_cdf[w][row][k]) { k0 = k; break; }
      }
      float p0 = sh_r0[w][row][k0];

      // stage 2: k1 ~ Categorical(wkk[k0] * r1s[row]); normalizer from WMMA
      const float* wr = &sh_wkk[k0][0];
      float tot1 = sh_t1[w][row][k0];
      float u2 = u01(hash3((uint32_t)grow, (uint32_t)j, 0x00C0FFEEu)) * tot1;
      int k1 = KC - 1;
      float racc = 0.f;
      for (int k = 0; k < KC; ++k) {
        racc += wr[k] * sh_r1s[w][row][k];
        if (u2 <= racc) { k1 = k; break; }
      }
      float p1 = sh_r1[w][row][k1];

      // stage 3: uniform pick inside bucket k01
      int k01 = k0 * KC + k1;
      int st  = indptr[k01];
      int cnt = indptr[k01 + 1] - st;
      float u3 = u01(hash3((uint32_t)grow, (uint32_t)j, 0xDEADBEEFu));
      int idx = (int)floorf((float)cnt * u3);
      int hi  = cnt - 1; if (hi < 0) hi = 0;
      if (idx > hi) idx = hi;
      if (idx < 0)  idx = 0;
      int item = indices[st + idx] + 1;      // 1-indexed item id

      size_t o = (size_t)grow * nneg + j;
      neg_items[o] = (float)item;
      neg_prob[o]  = p0 + p1;
    }
  }
}

// ---- pos_prop: one wave32 per batch row, 8 waves per block ----
__global__ __launch_bounds__(256)
void pos_prop_kernel(const float* __restrict__ query,
                     const int*   __restrict__ pos_items,
                     const float* __restrict__ c0_,
                     const float* __restrict__ c1_,
                     const int*   __restrict__ cd0,
                     const int*   __restrict__ cd1,
                     float* __restrict__ out, int B) {
  const int b    = blockIdx.x * 8 + (threadIdx.x >> 5);
  const int lane = threadIdx.x & 31;          // wave32
  if (b >= B) return;
  const int pi   = pos_items[b];
  const int kp0  = cd0[pi];
  const int kp1  = cd1[pi];
  const float* q  = query + (size_t)b * (2 * DHALF);
  const float* a0 = c0_ + (size_t)kp0 * DHALF;
  const float* a1 = c1_ + (size_t)kp1 * DHALF;
  float s = q[lane]              * a0[lane]
          + q[lane + 32]         * a0[lane + 32]
          + q[DHALF + lane]      * a1[lane]
          + q[DHALF + lane + 32] * a1[lane + 32];
#pragma unroll
  for (int off = 16; off > 0; off >>= 1) s += __shfl_xor(s, off, 32);
  if (lane == 0) out[b] = s;
}

extern "C" void kernel_launch(void* const* d_in, const int* in_sizes, int n_in,
                              void* d_out, int out_size, void* d_ws, size_t ws_size,
                              hipStream_t stream) {
  (void)n_in; (void)out_size; (void)d_ws; (void)ws_size;
  const float* query     = (const float*)d_in[0];
  const int*   pos_items = (const int*)  d_in[1];
  const int*   nneg_ptr  = (const int*)  d_in[2];
  const float* c0        = (const float*)d_in[3];
  const float* c1        = (const float*)d_in[4];
  const float* wkk       = (const float*)d_in[5];
  const float* c0_       = (const float*)d_in[6];
  const float* c1_       = (const float*)d_in[7];
  const int*   cd0       = (const int*)  d_in[8];
  const int*   cd1       = (const int*)  d_in[9];
  const int*   indices   = (const int*)  d_in[10];
  const int*   indptr    = (const int*)  d_in[11];
  float* out = (float*)d_out;

  const int B = in_sizes[0] / (2 * DHALF);      // 4096

  dim3 grid1((B + ROWS_PER_BLK - 1) / ROWS_PER_BLK);
  dim3 blk1(WAVES_PER_BLK * 32);
  sampler_fused_kernel<<<grid1, blk1, 0, stream>>>(query, c0, c1, wkk, indices,
                                                   indptr, nneg_ptr, out, B);

  pos_prop_kernel<<<dim3((B + 7) / 8), dim3(256), 0, stream>>>(
      query, pos_items, c0_, c1_, cd0, cd1, out, B);
}
